// ParallelRetention_37795712205445
// MI455X (gfx1250) — matrix operand order repaired and checked by reading, not compiled, not run on hardware
//
#include <hip/hip_runtime.h>
#include <math.h>

typedef __bf16 bf16;
typedef __attribute__((ext_vector_type(16))) __bf16 v16bf;
typedef __attribute__((ext_vector_type(8)))  __bf16 v8bf;
typedef __attribute__((ext_vector_type(8)))  float  v8f;
typedef __attribute__((ext_vector_type(4)))  unsigned u32x4;
typedef __attribute__((ext_vector_type(4)))  int      i32x4;
typedef __attribute__((ext_vector_type(8)))  int      i32x8;

#define WMMA_BF16(a,b,c) __builtin_amdgcn_wmma_f32_16x16x32_bf16(false,(a),false,(b),(short)0,(c),false,false)

static __device__ __forceinline__ v16bf cat8(v8bf lo, v8bf hi) {
  return __builtin_shufflevector(lo, hi, 0,1,2,3,4,5,6,7,8,9,10,11,12,13,14,15);
}
// A fragment from row-major LDS tile with row stride `ld`. Lane = 16*kh + m.
// elems 0..7 -> K = kh*8 + e ; elems 8..15 -> K = 16 + kh*8 + (e-8)   (ISA 7.12.2)
static __device__ __forceinline__ v16bf frag_a_ld(const bf16* t, int mRow, int kBase,
                                                  int ld, int lane) {
  int r = lane & 15, kh = lane >> 4;
  const bf16* base = &t[(size_t)(mRow + r) * ld + kBase];
  v8bf lo = *(const v8bf*)&base[kh * 8];
  v8bf hi = *(const v8bf*)&base[16 + kh * 8];
  return cat8(lo, hi);
}
static __device__ __forceinline__ v16bf frag_a(const bf16* t, int mRow, int lane) {
  return frag_a_ld(t, mRow, 0, 32, lane);
}
// B fragment: B[k][n] = W[n][k] (NT). Lane = 16*kh + n, elems 0..7 -> K = kh*8 + e ; elems 8..15 -> K = 16 + kh*8 + (e-8).
static __device__ __forceinline__ v16bf frag_b(const bf16* t, int nCol, int lane) {
  int c = lane & 15, kh = lane >> 4;
  const bf16* base = &t[(nCol + c) * 32];
  v8bf lo = *(const v8bf*)&base[kh * 8];
  v8bf hi = *(const v8bf*)&base[16 + kh * 8];
  return cat8(lo, hi);
}

// ---------------- Tensor Data Mover: 2-D tile -> LDS ----------------
// D# group0/group1 per CDNA5 ISA ch.8: count=1, type=2, data_size=1 (2 bytes),
// tile = tile_k x tile_rows, row stride = stride_elems.
static __device__ __forceinline__ void tdm_issue(u32x4 g0, i32x8 g1) {
  i32x4 z4 = {0, 0, 0, 0};
#if defined(__clang_major__) && __clang_major__ >= 23
  i32x8 z8 = {0, 0, 0, 0, 0, 0, 0, 0};
  __builtin_amdgcn_tensor_load_to_lds(g0, g1, z4, z4, z8, 0);
#else
  __builtin_amdgcn_tensor_load_to_lds(g0, g1, z4, z4, 0);
#endif
}
static __device__ __forceinline__ void tdm_load_tile_2d(
    const bf16* gtile, unsigned lds_byte_off, int tile_k, int tile_rows,
    int stride_elems) {
  unsigned long long ga = (unsigned long long)gtile;
  u32x4 g0;
  g0[0] = 1u;                                   // count=1 (valid descriptor)
  g0[1] = lds_byte_off;                         // lds_addr
  g0[2] = (unsigned)ga;                         // global_addr[31:0]
  g0[3] = (unsigned)((ga >> 32) & 0x1FFFFFFull) // global_addr[56:32]
          | (2u << 30);                         // type=2
  unsigned tk = (unsigned)tile_k, tr = (unsigned)tile_rows;
  unsigned st = (unsigned)stride_elems;
  i32x8 g1;
  g1[0] = (int)(1u << 16);                      // data_size=1 -> 2 bytes
  g1[1] = (int)((tk & 0xFFFFu) << 16);          // tensor_dim0[15:0]  @ bit 48
  g1[2] = (int)(((tk >> 16) & 0xFFFFu) | ((tr & 0xFFFFu) << 16)); // dim0 hi | dim1 lo
  g1[3] = (int)(((tr >> 16) & 0xFFFFu) | (tk << 16));             // dim1 hi | tile_dim0
  g1[4] = (int)tr;                              // tile_dim1 | tile_dim2=0
  g1[5] = (int)st;                              // tensor_dim0_stride[31:0]
  g1[6] = 0;
  g1[7] = 0;
  tdm_issue(g0, g1);
}

// ---------------- f32 -> bf16 convert ----------------
__global__ void k_cvt(const float* __restrict__ s, bf16* __restrict__ d, long long n) {
  long long i = (long long)blockIdx.x * blockDim.x + threadIdx.x;
  long long st = (long long)gridDim.x * blockDim.x;
  for (; i < n; i += st) d[i] = (bf16)s[i];
}

// ---------------- generic NT WMMA GEMM (TDM-fed, double-buffered) ----------------
// out = epi( A[M,K] @ Bw[Nc,K]^T + bias )
// EPI: 0 = bf16 row-major, 1 = bf16 transposed store (V^T), 2 = erf-GELU bf16,
//      3 = PReLU f32 (dual-A concat via A2/ksplit) + fused GroupNorm partial sums
template <int EPI>
__global__ __launch_bounds__(256) void gemm_nt(
    const bf16* __restrict__ A, const bf16* __restrict__ A2, int ksplit, int lda,
    const bf16* __restrict__ Bw, int ldb,
    const float* __restrict__ bias, const float* __restrict__ alpha,
    void* __restrict__ outp, int ldo, int Nc, int K,
    float* __restrict__ gsum, float* __restrict__ gsq) {
  (void)Nc;
  __shared__ alignas(32) bf16 As[2][128 * 32];
  __shared__ alignas(32) bf16 Bs[2][128 * 32];
  __shared__ float ssum[16], ssq[16];
  const int t = threadIdx.x, lane = t & 31, wid = t >> 5;
  const int m0 = blockIdx.y * 128, n0 = blockIdx.x * 128;
  const int wm = (wid & 1) * 64, wn = (wid >> 1) * 32;
  const v8f vz = {0.f, 0.f, 0.f, 0.f, 0.f, 0.f, 0.f, 0.f};
  v8f acc[4][2];
#pragma unroll
  for (int fm = 0; fm < 4; ++fm)
#pragma unroll
    for (int fn = 0; fn < 2; ++fn) acc[fm][fn] = vz;

  if (EPI == 3 && t < 16) { ssum[t] = 0.f; ssq[t] = 0.f; }

  const unsigned aOff[2] = {(unsigned)(size_t)(const void*)&As[0][0],
                            (unsigned)(size_t)(const void*)&As[1][0]};
  const unsigned bOff[2] = {(unsigned)(size_t)(const void*)&Bs[0][0],
                            (unsigned)(size_t)(const void*)&Bs[1][0]};

  auto issue = [&](int k0, int buf) {
    if (wid == 0) {
      const bf16* Ap = (A2 != nullptr && k0 >= ksplit) ? A2 : A;
      int kk = (A2 != nullptr && k0 >= ksplit) ? k0 - ksplit : k0;
      tdm_load_tile_2d(&Ap[(size_t)m0 * lda + kk], aOff[buf], 32, 128, lda);
    } else if (wid == 1) {
      tdm_load_tile_2d(&Bw[(size_t)n0 * ldb + k0], bOff[buf], 32, 128, ldb);
    }
  };

  issue(0, 0);
  if (wid < 2) __builtin_amdgcn_s_wait_tensorcnt(0);
  __syncthreads();

  const int nsteps = K / 32;
  for (int s = 0; s < nsteps; ++s) {
    const int buf = s & 1;
    if (s + 1 < nsteps) issue((s + 1) * 32, buf ^ 1);  // DMA next under compute
    const bf16* At = As[buf];
    const bf16* Bt = Bs[buf];
    v16bf af[4], bfr[2];
#pragma unroll
    for (int fm = 0; fm < 4; ++fm) af[fm] = frag_a(At, wm + fm * 16, lane);
#pragma unroll
    for (int fn = 0; fn < 2; ++fn) bfr[fn] = frag_b(Bt, wn + fn * 16, lane);
#pragma unroll
    for (int fm = 0; fm < 4; ++fm)
#pragma unroll
      for (int fn = 0; fn < 2; ++fn)
        acc[fm][fn] = WMMA_BF16(af[fm], bfr[fn], acc[fm][fn]);
    if (wid < 2) __builtin_amdgcn_s_wait_tensorcnt(0);
    __syncthreads();
  }
  // C layout: elem r of lane -> M = r + 8*(lane/16), N = lane%16 (ISA 7.12.2)
  const int colq = lane & 15, half = lane >> 4;
#pragma unroll
  for (int fm = 0; fm < 4; ++fm) {
#pragma unroll
    for (int fn = 0; fn < 2; ++fn) {
      v8f cc = acc[fm][fn];
      int gcol = n0 + wn + fn * 16 + colq;
      int grow0 = m0 + wm + fm * 16 + half * 8;
      float bv = bias ? bias[gcol] : 0.f;
      float av = (EPI == 3 && alpha) ? alpha[gcol] : 0.f;
      float ls = 0.f, lsq = 0.f;
#pragma unroll
      for (int r = 0; r < 8; ++r) {
        float v = cc[r] + bv;
        if (EPI == 2) v = 0.5f * v * (1.f + erff(v * 0.70710678118654752f));
        if (EPI == 3) v = (v >= 0.f) ? v : av * v;
        size_t row = (size_t)(grow0 + r);
        if (EPI == 1)
          ((bf16*)outp)[(size_t)gcol * ldo + row] = (bf16)v;
        else if (EPI == 3) {
          ((float*)outp)[row * (size_t)ldo + gcol] = v;
          ls += v;
          lsq += v * v;
        } else
          ((bf16*)outp)[row * (size_t)ldo + gcol] = (bf16)v;
      }
      if (EPI == 3) {  // fused GroupNorm partials (group = col/32, uniform per frag)
        atomicAdd(&ssum[(gcol >> 5) & 15], ls);
        atomicAdd(&ssq[(gcol >> 5) & 15], lsq);
      }
    }
  }
  if (EPI == 3) {
    __syncthreads();
    if (t < 16) {
      atomicAdd(&gsum[t], ssum[t]);
      atomicAdd(&gsq[t], ssq[t]);
    }
  }
}

// ---------------- attention passes ----------------
// Q block (128 x 512 bf16 = 128 KB) resident in LDS via one TDM descriptor;
// K tiles (64 x 32) streamed double-buffered by TDM.
// PHASE 0: online rowmax / rowsum of t = D * (Q.K^T) * scale
// PHASE 1: P = exp(t - rowmax) / rowsum  (bf16)
template <int PHASE>
__global__ __launch_bounds__(256) void attn_pass(
    const bf16* __restrict__ Q, const bf16* __restrict__ Kb,
    const float* __restrict__ Dm, float* __restrict__ rowmax,
    float* __restrict__ rowsum, bf16* __restrict__ P, int N, int dK) {
  __shared__ alignas(32) bf16 Qs[128 * 512];   // persistent Q block (128 KB)
  __shared__ alignas(32) bf16 Ks[2][64 * 32];  // double-buffered K tiles
  const int t = threadIdx.x, lane = t & 31, wid = t >> 5;
  const int m0 = blockIdx.x * 128;
  const int half = lane >> 4;
  const int rbase = m0 + wid * 16 + half * 8;  // 8 rows owned by this lane
  const float scale = 0.044194173824159216f;   // 1/sqrt(512)
  const unsigned qOff = (unsigned)(size_t)(const void*)&Qs[0];
  const unsigned kOff[2] = {(unsigned)(size_t)(const void*)&Ks[0][0],
                            (unsigned)(size_t)(const void*)&Ks[1][0]};
  float mrow[8], lrow[8];
#pragma unroll
  for (int r = 0; r < 8; ++r) {
    if (PHASE == 0) { mrow[r] = -1e30f; lrow[r] = 0.f; }
    else            { mrow[r] = rowmax[rbase + r]; lrow[r] = 1.f / rowsum[rbase + r]; }
  }
  const v8f vz = {0.f, 0.f, 0.f, 0.f, 0.f, 0.f, 0.f, 0.f};

  // prologue: whole Q block + first K tile
  if (wid == 0)
    tdm_load_tile_2d(&Q[(size_t)m0 * dK], qOff, dK, 128, dK);
  else if (wid == 1)
    tdm_load_tile_2d(&Kb[0], kOff[0], 32, 64, dK);
  if (wid < 2) __builtin_amdgcn_s_wait_tensorcnt(0);
  __syncthreads();

  const int totalSteps = (N / 64) * (dK / 32);
  int step = 0;
  for (int jt = 0; jt < N; jt += 64) {
    v8f s[4] = {vz, vz, vz, vz};
    for (int ks = 0; ks < dK; ks += 32, ++step) {
      const int buf = step & 1;
      if (wid == 1 && step + 1 < totalSteps) {  // stream next K tile under compute
        int njt = (ks + 32 < dK) ? jt : jt + 64;
        int nks = (ks + 32 < dK) ? ks + 32 : 0;
        tdm_load_tile_2d(&Kb[(size_t)njt * dK + nks], kOff[buf ^ 1], 32, 64, dK);
      }
      v16bf a = frag_a_ld(Qs, wid * 16, ks, dK, lane);
#pragma unroll
      for (int f = 0; f < 4; ++f) {
        v16bf b = frag_b(Ks[buf], f * 16, lane);
        s[f] = WMMA_BF16(a, b, s[f]);
      }
      if (wid == 1) __builtin_amdgcn_s_wait_tensorcnt(0);
      __syncthreads();
    }
    if (jt + 64 < N)  // stream the only HBM-resident operand ahead
      __builtin_prefetch(&Dm[(size_t)rbase * N + jt + 64 + lane], 0, 1);

    const int colb = jt + (lane & 15);
    float tv[4][8];
#pragma unroll
    for (int f = 0; f < 4; ++f)
#pragma unroll
      for (int r = 0; r < 8; ++r)
        tv[f][r] = Dm[(size_t)(rbase + r) * N + colb + f * 16] * (s[f][r] * scale);

    if (PHASE == 0) {
#pragma unroll
      for (int r = 0; r < 8; ++r) {
        float mx = fmaxf(fmaxf(tv[0][r], tv[1][r]), fmaxf(tv[2][r], tv[3][r]));
        mx = fmaxf(mx, __shfl_xor(mx, 1));
        mx = fmaxf(mx, __shfl_xor(mx, 2));
        mx = fmaxf(mx, __shfl_xor(mx, 4));
        mx = fmaxf(mx, __shfl_xor(mx, 8));  // stays inside 16-lane half
        float mnew = fmaxf(mrow[r], mx);
        float se = __expf(tv[0][r] - mnew) + __expf(tv[1][r] - mnew) +
                   __expf(tv[2][r] - mnew) + __expf(tv[3][r] - mnew);
        se += __shfl_xor(se, 1);
        se += __shfl_xor(se, 2);
        se += __shfl_xor(se, 4);
        se += __shfl_xor(se, 8);
        lrow[r] = lrow[r] * __expf(mrow[r] - mnew) + se;
        mrow[r] = mnew;
      }
    } else {
#pragma unroll
      for (int f = 0; f < 4; ++f)
#pragma unroll
        for (int r = 0; r < 8; ++r)
          P[(size_t)(rbase + r) * N + colb + f * 16] =
              (bf16)(__expf(tv[f][r] - mrow[r]) * lrow[r]);
    }
  }
  if (PHASE == 0 && (lane & 15) == 0) {
#pragma unroll
    for (int r = 0; r < 8; ++r) {
      rowmax[rbase + r] = mrow[r];
      rowsum[rbase + r] = lrow[r];
    }
  }
}

// ---------------- GroupNorm finalize ----------------
__global__ void k_gn_zero(float* g) {
  if (threadIdx.x < 64) g[threadIdx.x] = 0.f;
}
__global__ void k_gn_apply(const float* __restrict__ x, const float* __restrict__ gsum,
                           const float* __restrict__ gsq, const float* __restrict__ gamma,
                           const float* __restrict__ beta, float* __restrict__ out,
                           size_t total) {
  const float inv_cnt = 1.f / (32.f * 8192.f);
  size_t i = (size_t)blockIdx.x * blockDim.x + threadIdx.x;
  size_t st = (size_t)gridDim.x * blockDim.x;
  for (; i < total; i += st) {
    int c = (int)(i & 511);
    int g = c >> 5;
    float mean = gsum[g] * inv_cnt;
    float var = gsq[g] * inv_cnt - mean * mean;
    out[i] = (x[i] - mean) * rsqrtf(var + 1e-5f) * gamma[c] + beta[c];
  }
}

// ---------------- host orchestration ----------------
extern "C" void kernel_launch(void* const* d_in, const int* in_sizes, int n_in,
                              void* d_out, int out_size, void* d_ws, size_t ws_size,
                              hipStream_t stream) {
  (void)in_sizes; (void)n_in; (void)out_size; (void)ws_size;
  const float* h  = (const float*)d_in[0];
  const float* Dm = (const float*)d_in[1];
  const float* hp = (const float*)d_in[2];
  const float* Wq = (const float*)d_in[3];  const float* bq = (const float*)d_in[4];
  const float* Wk = (const float*)d_in[5];  const float* bk = (const float*)d_in[6];
  const float* Wv = (const float*)d_in[7];  const float* bv = (const float*)d_in[8];
  const float* Wr = (const float*)d_in[9];  const float* br = (const float*)d_in[10];
  const float* Wc = (const float*)d_in[11]; const float* bc = (const float*)d_in[12];
  const float* Wp = (const float*)d_in[13]; const float* bp = (const float*)d_in[14];
  const float* al = (const float*)d_in[15];
  const float* ga = (const float*)d_in[16];
  const float* be = (const float*)d_in[17];

  const int N = 8192, DIN = 512, INT_ = 512, HID_ = 1024, OUT_ = 512;

  char* ws = (char*)d_ws;
  size_t off = 0;
  auto alloc = [&](size_t bytes) -> void* {
    void* p = (void*)(ws + off);
    off += (bytes + 255) & ~(size_t)255;
    return p;
  };
  bf16* hB   = (bf16*)alloc((size_t)N * DIN * 2);
  bf16* hpB  = (bf16*)alloc((size_t)N * OUT_ * 2);
  bf16* WqB  = (bf16*)alloc((size_t)INT_ * DIN * 2);
  bf16* WkB  = (bf16*)alloc((size_t)INT_ * DIN * 2);
  bf16* WvB  = (bf16*)alloc((size_t)INT_ * DIN * 2);
  bf16* WrB  = (bf16*)alloc((size_t)HID_ * INT_ * 2);
  bf16* WcB  = (bf16*)alloc((size_t)HID_ * OUT_ * 2);
  bf16* WpB  = (bf16*)alloc((size_t)OUT_ * 2 * HID_ * 2);
  bf16* Qb   = (bf16*)alloc((size_t)N * INT_ * 2);
  bf16* KbB  = (bf16*)alloc((size_t)N * INT_ * 2);
  bf16* VtB  = (bf16*)alloc((size_t)INT_ * N * 2);  // V transposed
  float* rmx = (float*)alloc((size_t)N * 4);
  float* rsm = (float*)alloc((size_t)N * 4);
  bf16* Pb   = (bf16*)alloc((size_t)N * N * 2);     // softmax probs
  bf16* hrB  = (bf16*)alloc((size_t)N * INT_ * 2);
  bf16* rB   = (bf16*)alloc((size_t)N * HID_ * 2);
  bf16* cB   = (bf16*)alloc((size_t)N * HID_ * 2);
  float* xB  = (float*)alloc((size_t)N * OUT_ * 4);
  float* gst = (float*)alloc(64 * 4);  // gsum[16] | gsq[16]

  dim3 blk(256);
  auto cvt = [&](const float* s, bf16* d, size_t n) {
    k_cvt<<<dim3(512), blk, 0, stream>>>(s, d, (long long)n);
  };
  cvt(h, hB, (size_t)N * DIN);
  cvt(hp, hpB, (size_t)N * OUT_);
  cvt(Wq, WqB, (size_t)INT_ * DIN);
  cvt(Wk, WkB, (size_t)INT_ * DIN);
  cvt(Wv, WvB, (size_t)INT_ * DIN);
  cvt(Wr, WrB, (size_t)HID_ * INT_);
  cvt(Wc, WcB, (size_t)HID_ * OUT_);
  cvt(Wp, WpB, (size_t)OUT_ * 2 * HID_);

  const int KBIG = 1 << 30;
  // Q, K, V^T
  gemm_nt<0><<<dim3(INT_ / 128, N / 128), blk, 0, stream>>>(
      hB, nullptr, KBIG, DIN, WqB, DIN, bq, nullptr, Qb, INT_, INT_, DIN, nullptr, nullptr);
  gemm_nt<0><<<dim3(INT_ / 128, N / 128), blk, 0, stream>>>(
      hB, nullptr, KBIG, DIN, WkB, DIN, bk, nullptr, KbB, INT_, INT_, DIN, nullptr, nullptr);
  gemm_nt<1><<<dim3(INT_ / 128, N / 128), blk, 0, stream>>>(
      hB, nullptr, KBIG, DIN, WvB, DIN, bv, nullptr, VtB, N, INT_, DIN, nullptr, nullptr);
  // softmax stats + probability materialization
  attn_pass<0><<<dim3(N / 128), blk, 0, stream>>>(Qb, KbB, Dm, rmx, rsm, nullptr, N, INT_);
  attn_pass<1><<<dim3(N / 128), blk, 0, stream>>>(Qb, KbB, Dm, rmx, rsm, Pb, N, INT_);
  // h_ret = P @ V   (B = V^T rows = channels)
  gemm_nt<0><<<dim3(INT_ / 128, N / 128), blk, 0, stream>>>(
      Pb, nullptr, KBIG, N, VtB, N, nullptr, nullptr, hrB, INT_, INT_, N, nullptr, nullptr);
  // r = gelu(h_ret @ Wr^T + br) ; c = h_prime @ Wc^T + bc
  gemm_nt<2><<<dim3(HID_ / 128, N / 128), blk, 0, stream>>>(
      hrB, nullptr, KBIG, INT_, WrB, INT_, br, nullptr, rB, HID_, HID_, INT_, nullptr, nullptr);
  gemm_nt<0><<<dim3(HID_ / 128, N / 128), blk, 0, stream>>>(
      hpB, nullptr, KBIG, OUT_, WcB, OUT_, bc, nullptr, cB, HID_, HID_, OUT_, nullptr, nullptr);
  // x = prelu([r|c] @ Wp^T + bp) with fused GroupNorm partial sums
  k_gn_zero<<<dim3(1), dim3(64), 0, stream>>>(gst);
  gemm_nt<3><<<dim3(OUT_ / 128, N / 128), blk, 0, stream>>>(
      rB, cB, HID_, HID_, WpB, 2 * HID_, bp, al, xB, OUT_, OUT_, 2 * HID_, gst, gst + 16);
  // GroupNorm finalize (16 groups over channels x N)
  k_gn_apply<<<dim3(2048), blk, 0, stream>>>(xB, gst, gst + 16, ga, be,
                                             (float*)d_out, (size_t)N * OUT_);
}
